// GRU_79937931313743
// MI455X (gfx1250) — matrix operand order repaired and checked by reading, compiled, not run on hardware
//
#include <hip/hip_runtime.h>
#include <hip/hip_bf16.h>

typedef __attribute__((ext_vector_type(16))) _Float16 v16h;
typedef __attribute__((ext_vector_type(8)))  _Float16 v8h;
typedef __attribute__((ext_vector_type(8)))  float    v8f;

#define S_  64
#define B_  64
#define H_  512
#define E_  512
#define L_  2
#define V_  32000
#define KG  1024               // E_ + H_ (GRU GEMM reduction dim)

// ---------------------------------------------------------------------------
// WMMA helpers (CDNA5 wave32, v_wmma_f32_16x16x32_f16)
// ---------------------------------------------------------------------------
__device__ __forceinline__ v8f wmma_16x16x32(v16h a, v16h b, v8f c) {
  // (neg_a, A, neg_b, B, c_mod, C, reuse_a, reuse_b)
  return __builtin_amdgcn_wmma_f32_16x16x32_f16(false, a, false, b, (short)0, c,
                                                false, false);
}

__device__ __forceinline__ v16h cat16(v8h lo, v8h hi) {
  return __builtin_shufflevector(lo, hi, 0, 1, 2, 3, 4, 5, 6, 7, 8, 9, 10, 11,
                                 12, 13, 14, 15);
}

// A fragment from a row-major f16 matrix with row stride 512.
// 16-bit A layout (16x32, M = lane&15), contiguous K to match B:
//   lane[4]==0: K = k0+0..15
//   lane[4]==1: K = k0+16..31
__device__ __forceinline__ v16h load_a512(const _Float16* __restrict__ base,
                                          int row, int k0, int hi) {
  const int ka = k0 + hi * 16;
  const _Float16* p = base + (size_t)row * H_ + ka;
  return cat16(*(const v8h*)p, *(const v8h*)(p + 8));
}

// ---------------------------------------------------------------------------
// Tensor Data Mover: contiguous 1-D block copy global -> LDS.
// D# built per CDNA5 ISA 8.3/8.4 (group0: count/lds_addr/global_addr/type=2;
// group1: data_size=2B, dim0 = tile0 = nelem, stride = nelem).
// This toolchain exposes the 6-arg builtin:
//   (u32x4 g0, i32x8 g1, i32x4 g2, i32x4 g3, i32x8 extra, i32 cpol)
// ---------------------------------------------------------------------------
#if __has_builtin(__builtin_amdgcn_tensor_load_to_lds)
#define HAVE_TDM 1
typedef __attribute__((ext_vector_type(4))) unsigned int u32x4;
typedef __attribute__((ext_vector_type(8))) int i32x8;
typedef __attribute__((ext_vector_type(4))) int i32x4;

__device__ __forceinline__ void tdm_copy_f16_to_lds(unsigned lds_off,
                                                    const void* gaddr,
                                                    unsigned nelem) {
  const unsigned long long ga = (unsigned long long)gaddr;
  u32x4 g0;
  g0.x = 0x1u;                                   // count=1 valid descriptor
  g0.y = lds_off;                                // lds_addr
  g0.z = (unsigned)(ga & 0xffffffffu);           // global_addr[31:0]
  g0.w = (unsigned)((ga >> 32) & 0x1ffffffu)     // global_addr[56:32]
         | (2u << 30);                           // type=2 ("image")
  i32x8 g1;
  g1[0] = (int)(1u << 16);                       // wg_mask=0, data_size=1 (2B)
  g1[1] = (int)((nelem & 0xffffu) << 16);        // tensor_dim0[15:0]
  g1[2] = (int)(((nelem >> 16) & 0xffffu) |      // tensor_dim0[31:16]
                (1u << 16));                     // tensor_dim1 = 1
  g1[3] = (int)((nelem & 0xffffu) << 16);        // tile_dim0 (<=65535)
  g1[4] = 1;                                     // tile_dim1=1, tile_dim2=0
  g1[5] = (int)nelem;                            // tensor_dim0_stride[31:0]
  g1[6] = 0;
  g1[7] = 0;
  const i32x4 z4 = {0, 0, 0, 0};
  const i32x8 z8 = {0, 0, 0, 0, 0, 0, 0, 0};
  __builtin_amdgcn_tensor_load_to_lds(g0, g1, z4, z4, z8, 0);
}
#else
#define HAVE_TDM 0
#endif

// ---------------------------------------------------------------------------
// Prep kernels
// ---------------------------------------------------------------------------
// src is [K][N] f32, dst is [N][K] f16 (transposed, contiguous along K)
__global__ __launch_bounds__(256) void transpose_cvt_kernel(
    const float* __restrict__ src, _Float16* __restrict__ dst, int K, int N) {
  const long total = (long)K * N;
  for (long i = (long)blockIdx.x * blockDim.x + threadIdx.x; i < total;
       i += (long)gridDim.x * blockDim.x) {
    const long n = i / K;
    const long k = i - n * K;
    dst[i] = (_Float16)src[k * (long)N + n];
  }
}

__global__ __launch_bounds__(256) void gather_emb_kernel(
    const int* __restrict__ tokens, const float* __restrict__ emb,
    _Float16* __restrict__ x) {
  const long total = (long)S_ * B_ * E_;
  for (long i = (long)blockIdx.x * blockDim.x + threadIdx.x; i < total;
       i += (long)gridDim.x * blockDim.x) {
    const long row = i / E_;          // s*B + b
    const long e = i - row * E_;
    x[i] = (_Float16)emb[(long)tokens[row] * E_ + e];
  }
}

__global__ __launch_bounds__(256) void init_h_kernel(
    const float* __restrict__ h_in, float* __restrict__ h32,
    _Float16* __restrict__ h16) {
  const int i = blockIdx.x * blockDim.x + threadIdx.x;
  if (i < L_ * B_ * H_) {
    const float v = h_in[i];
    h32[i] = v;
    h16[i] = (_Float16)v;
  }
}

__global__ __launch_bounds__(256) void copy_h_kernel(
    const float* __restrict__ h32, float* __restrict__ dst) {
  const int i = blockIdx.x * blockDim.x + threadIdx.x;
  if (i < L_ * B_ * H_) dst[i] = h32[i];
}

// ---------------------------------------------------------------------------
// GRU step, part 1: r/z gates.  M=64, N=512, K=1024 (split at 512 so each
// half-loop has a fixed base pointer -> no per-iteration address selects).
// 16 blocks x 8 waves, one 16x16 tile per wave, dual accumulators (r,z).
// ---------------------------------------------------------------------------
__global__ __launch_bounds__(256) void gru_rz_kernel(
    const _Float16* __restrict__ inp, const _Float16* __restrict__ hprev16,
    const _Float16* __restrict__ WrT, const _Float16* __restrict__ WzT,
    const float* __restrict__ br, const float* __restrict__ bz,
    const float* __restrict__ hprev32, _Float16* __restrict__ rh,
    float* __restrict__ z_out) {
  const int tid = threadIdx.x;
  const int lane = tid & 31;
  const int wave = tid >> 5;
  const int l15 = lane & 15;
  const int hi = lane >> 4;
  const int tile = blockIdx.x * 8 + wave;   // 0..127
  const int m0 = (tile >> 5) << 4;          // batch tile
  const int n0 = (tile & 31) << 4;          // hidden-unit tile

  v8f accR = {};
  v8f accZ = {};
  const int arow = m0 + l15;
  const _Float16* wr = WrT + (size_t)(n0 + l15) * KG;
  const _Float16* wz = WzT + (size_t)(n0 + l15) * KG;
#pragma unroll 4
  for (int k0 = 0; k0 < H_; k0 += 32) {     // K in [0,512): A = inp
    const v16h a = load_a512(inp, arow, k0, hi);
    const v16h bR = *(const v16h*)(wr + k0 + hi * 16);
    const v16h bZ = *(const v16h*)(wz + k0 + hi * 16);
    accR = wmma_16x16x32(a, bR, accR);
    accZ = wmma_16x16x32(a, bZ, accZ);
  }
#pragma unroll 4
  for (int k0 = 0; k0 < H_; k0 += 32) {     // K in [512,1024): A = h_prev
    const v16h a = load_a512(hprev16, arow, k0, hi);
    const v16h bR = *(const v16h*)(wr + H_ + k0 + hi * 16);
    const v16h bZ = *(const v16h*)(wz + H_ + k0 + hi * 16);
    accR = wmma_16x16x32(a, bR, accR);
    accZ = wmma_16x16x32(a, bZ, accZ);
  }
  const int n = n0 + l15;
  const float brv = br[n];
  const float bzv = bz[n];
#pragma unroll
  for (int j = 0; j < 8; ++j) {
    const int b = m0 + hi * 8 + j;          // C layout: M = j + 8*lane[4]
    const size_t idx = (size_t)b * H_ + n;
    const float r = 1.0f / (1.0f + __expf(-(accR[j] + brv)));
    const float z = 1.0f / (1.0f + __expf(-(accZ[j] + bzv)));
    rh[idx] = (_Float16)(r * hprev32[idx]);
    z_out[idx] = z;
  }
}

// ---------------------------------------------------------------------------
// GRU step, part 2: candidate + state blend.
// ---------------------------------------------------------------------------
__global__ __launch_bounds__(256) void gru_h_kernel(
    const _Float16* __restrict__ inp, const _Float16* __restrict__ rh,
    const _Float16* __restrict__ WhT, const float* __restrict__ bh,
    const float* __restrict__ z_in, float* __restrict__ h32,
    _Float16* __restrict__ h16, _Float16* __restrict__ ydst) {
  const int tid = threadIdx.x;
  const int lane = tid & 31;
  const int wave = tid >> 5;
  const int l15 = lane & 15;
  const int hi = lane >> 4;
  const int tile = blockIdx.x * 8 + wave;
  const int m0 = (tile >> 5) << 4;
  const int n0 = (tile & 31) << 4;

  v8f acc = {};
  const int arow = m0 + l15;
  const _Float16* wh = WhT + (size_t)(n0 + l15) * KG;
#pragma unroll 4
  for (int k0 = 0; k0 < H_; k0 += 32) {     // K in [0,512): A = inp
    const v16h a = load_a512(inp, arow, k0, hi);
    const v16h bH = *(const v16h*)(wh + k0 + hi * 16);
    acc = wmma_16x16x32(a, bH, acc);
  }
#pragma unroll 4
  for (int k0 = 0; k0 < H_; k0 += 32) {     // K in [512,1024): A = r*h
    const v16h a = load_a512(rh, arow, k0, hi);
    const v16h bH = *(const v16h*)(wh + H_ + k0 + hi * 16);
    acc = wmma_16x16x32(a, bH, acc);
  }
  const int n = n0 + l15;
  const float bhv = bh[n];
#pragma unroll
  for (int j = 0; j < 8; ++j) {
    const int b = m0 + hi * 8 + j;
    const size_t idx = (size_t)b * H_ + n;
    const float hh = tanhf(acc[j] + bhv);
    const float z = z_in[idx];
    const float nh = (1.0f - z) * h32[idx] + z * hh;
    h32[idx] = nh;
    h16[idx] = (_Float16)nh;
    if (ydst) ydst[idx] = (_Float16)nh;
  }
}

// ---------------------------------------------------------------------------
// Final projection: [4096, 512] x [512, 32000] (+bout).
// Block = 8 waves (2 M x 4 N), block tile 64x256, wave tile 32x64
// (two 16-row subtiles share each B fragment -> 8 wmma per 4 B loads).
// A-slab (64x512 f16 = 64KB, contiguous) staged in LDS via TDM.
// ---------------------------------------------------------------------------
__global__ __launch_bounds__(256) void logits_kernel(
    const _Float16* __restrict__ Y, const _Float16* __restrict__ WoutT,
    const float* __restrict__ bout, float* __restrict__ out) {
  __shared__ __align__(16) _Float16 As[64 * 512];
  const int tid = threadIdx.x;
  const int lane = tid & 31;
  const int wave = tid >> 5;
  const int l15 = lane & 15;
  const int hi = lane >> 4;
  const int wm = wave >> 2;                 // 0..1 -> M rows [wm*32, +32)
  const int wn = wave & 3;                  // 0..3 -> N cols [wn*64, +64)
  const int m_base = blockIdx.y * 64;       // 64 blocks in M
  const int n_base = blockIdx.x * 256;      // 125 blocks in N

  // Stage contiguous 64-row A slab into LDS (rows of Y are contiguous).
#if HAVE_TDM
  if (wave == 0) {
    tdm_copy_f16_to_lds((unsigned)(size_t)(void*)As,
                        Y + (size_t)m_base * 512, 64 * 512);
    __builtin_amdgcn_s_wait_tensorcnt(0);
  }
#else
  {
    const v8h* src = (const v8h*)(Y + (size_t)m_base * 512);
    v8h* dst = (v8h*)As;
#pragma unroll
    for (int i = tid; i < 4096; i += 256) dst[i] = src[i];
  }
#endif
  __syncthreads();

  v8f acc[2][4] = {};
  const int ar0 = wm * 32 + l15;            // M subtile 0 row
  const int ar1 = ar0 + 16;                 // M subtile 1 row
  const _Float16* wbase = WoutT + (size_t)(n_base + wn * 64 + l15) * 512;
#pragma unroll 2
  for (int k0 = 0; k0 < 512; k0 += 32) {
    const int ka = k0 + hi * 16;
    const v16h a0 = cat16(*(const v8h*)&As[ar0 * 512 + ka],
                          *(const v8h*)&As[ar0 * 512 + ka + 8]);
    const v16h a1 = cat16(*(const v8h*)&As[ar1 * 512 + ka],
                          *(const v8h*)&As[ar1 * 512 + ka + 8]);
    const int kb = k0 + hi * 16;
#pragma unroll
    for (int sub = 0; sub < 4; ++sub) {
      const v16h b = *(const v16h*)(wbase + (size_t)sub * 16 * 512 + kb);
      acc[0][sub] = wmma_16x16x32(a0, b, acc[0][sub]);
      acc[1][sub] = wmma_16x16x32(a1, b, acc[1][sub]);
    }
  }

#pragma unroll
  for (int sm = 0; sm < 2; ++sm) {
#pragma unroll
    for (int sub = 0; sub < 4; ++sub) {
      const int n = n_base + wn * 64 + sub * 16 + l15;
      const float bv = bout[n];
#pragma unroll
      for (int j = 0; j < 8; ++j) {
        const int m = m_base + wm * 32 + sm * 16 + hi * 8 + j;
        out[(size_t)m * V_ + n] = acc[sm][sub][j] + bv;
      }
    }
  }
}

// ---------------------------------------------------------------------------
// Host driver
// ---------------------------------------------------------------------------
extern "C" void kernel_launch(void* const* d_in, const int* in_sizes, int n_in,
                              void* d_out, int out_size, void* d_ws,
                              size_t ws_size, hipStream_t stream) {
  const int* tokens = (const int*)d_in[0];
  const float* hidden0 = (const float*)d_in[1];
  const float* emb = (const float*)d_in[2];
  const float* Wr = (const float*)d_in[3];
  const float* br = (const float*)d_in[4];
  const float* Wz = (const float*)d_in[5];
  const float* bz = (const float*)d_in[6];
  const float* Wh = (const float*)d_in[7];
  const float* bh = (const float*)d_in[8];
  const float* Wout = (const float*)d_in[9];
  const float* bout = (const float*)d_in[10];
  float* out = (float*)d_out;

  (void)in_sizes; (void)n_in; (void)out_size; (void)ws_size;

  // Workspace carve-up (all hipMalloc'd base => well aligned).
  char* ws = (char*)d_ws;
  size_t off = 0;
  auto carve = [&](size_t bytes) -> void* {
    void* p = ws + off;
    off = (off + bytes + 255) & ~(size_t)255;
    return p;
  };
  _Float16* WrT = (_Float16*)carve((size_t)L_ * H_ * KG * 2);
  _Float16* WzT = (_Float16*)carve((size_t)L_ * H_ * KG * 2);
  _Float16* WhT = (_Float16*)carve((size_t)L_ * H_ * KG * 2);
  _Float16* WoutT = (_Float16*)carve((size_t)V_ * H_ * 2);
  _Float16* xall = (_Float16*)carve((size_t)S_ * B_ * E_ * 2);
  _Float16* y16 = (_Float16*)carve((size_t)S_ * B_ * H_ * 2);
  float* h32 = (float*)carve((size_t)L_ * B_ * H_ * 4);
  _Float16* h16 = (_Float16*)carve((size_t)L_ * B_ * H_ * 2);
  _Float16* rh16 = (_Float16*)carve((size_t)L_ * B_ * H_ * 2);
  float* z32 = (float*)carve((size_t)L_ * B_ * H_ * 4);

  // --- Prep: weight transpose+f16 conversion, embedding gather, h init ---
  for (int l = 0; l < L_; ++l) {
    const size_t wo = (size_t)l * KG * H_;
    transpose_cvt_kernel<<<512, 256, 0, stream>>>(Wr + wo, WrT + wo, KG, H_);
    transpose_cvt_kernel<<<512, 256, 0, stream>>>(Wz + wo, WzT + wo, KG, H_);
    transpose_cvt_kernel<<<512, 256, 0, stream>>>(Wh + wo, WhT + wo, KG, H_);
  }
  transpose_cvt_kernel<<<4096, 256, 0, stream>>>(Wout, WoutT, H_, V_);
  gather_emb_kernel<<<2048, 256, 0, stream>>>(tokens, emb, xall);
  init_h_kernel<<<(L_ * B_ * H_ + 255) / 256, 256, 0, stream>>>(hidden0, h32,
                                                                h16);

  // --- Sequential GRU over time ---
  for (int t = 0; t < S_; ++t) {
    for (int l = 0; l < L_; ++l) {
      const _Float16* inp =
          (l == 0) ? (xall + (size_t)t * B_ * E_) : h16;  // layer0's h16
      const size_t lo = (size_t)l * B_ * H_;
      const size_t lw = (size_t)l * H_ * KG;
      gru_rz_kernel<<<16, 256, 0, stream>>>(
          inp, h16 + lo, WrT + lw, WzT + lw, br + (size_t)l * H_,
          bz + (size_t)l * H_, h32 + lo, rh16 + lo, z32 + lo);
      _Float16* ydst = (l == L_ - 1) ? (y16 + (size_t)t * B_ * H_) : nullptr;
      gru_h_kernel<<<16, 256, 0, stream>>>(inp, rh16 + lo, WhT + lw,
                                           bh + (size_t)l * H_, z32 + lo,
                                           h32 + lo, h16 + lo, ydst);
    }
  }

  // --- Batched output projection over all timesteps ---
  dim3 lgrid(V_ / 256, (S_ * B_) / 64);  // 125 x 64
  logits_kernel<<<lgrid, 256, 0, stream>>>(y16, WoutT, bout, out);

  // --- Final hidden state to output tail ---
  copy_h_kernel<<<(L_ * B_ * H_ + 255) / 256, 256, 0, stream>>>(
      h32, out + (size_t)S_ * B_ * V_);
}
